// MS2D_46686294507588
// MI455X (gfx1250) — matrix operand order repaired and checked by reading, not compiled, not run on hardware
//
#include <hip/hip_runtime.h>
#include <hip/hip_bf16.h>

typedef __attribute__((ext_vector_type(16))) _Float16 v16h;
typedef __attribute__((ext_vector_type(8)))  float    v8f;

#define B_   4
#define H_   64
#define W_   64
#define L_   4096
#define DM_  96
#define DI_  96
#define NS_  16
#define K_   4
#define DTR_ 6
#define CD_  38
#define CDP_ 48   // 38 padded to 3 WMMA N-tiles
#define CH_  32   // scan chunks per chain
#define CL_  128  // steps per chunk (CH_*CL_ == L_)

// ---------------- WMMA operand loaders (f32 -> f16 convert on load) ----------
// A tile 16x32 f16: lane<16 holds row M=lane, K = 0..15; lane>=16 same
// rows, K = 16..31.
__device__ __forceinline__ v16h load_a16(const float* __restrict__ a_row, int k0, int lane) {
  const int off = (lane & 16) ? 16 : 0;
  const float* p = a_row + k0 + off;
  v16h r;
#pragma unroll
  for (int j = 0; j < 8; ++j) {
    r[j]     = (_Float16)p[j];
    r[j + 8] = (_Float16)p[j + 8];
  }
  return r;
}
// B tile 32x16 f16: N = lane&15; lane<16 holds K=k0..k0+15, lane>=16 K=k0+16..k0+31.
// Weight rows are (N, Kdim) row-major -> 16 contiguous floats per lane.
__device__ __forceinline__ v16h load_b16(const float* __restrict__ w_row, int k0, int lane, bool valid) {
  const int off = (lane & 16) ? 16 : 0;
  const float* p = w_row + k0 + off;
  v16h r;
#pragma unroll
  for (int j = 0; j < 16; ++j) r[j] = valid ? (_Float16)p[j] : (_Float16)0.0f;
  return r;
}

__device__ __forceinline__ float silu(float v) { return v / (1.0f + __expf(-v)); }

// ---------------- K1: in_proj GEMM: xz = x @ W^T, split into xc_pre / silu(z) -
__global__ void k_in_proj(const float* __restrict__ x, const float* __restrict__ w,
                          float* __restrict__ xc_pre, float* __restrict__ sz) {
  const int lane = threadIdx.x & 31;
  const int m0 = blockIdx.x * 16;
  const int n0 = blockIdx.y * 16;
  const int arow = m0 + (lane & 15);
  const int col  = n0 + (lane & 15);
  v8f acc = {};
#pragma unroll
  for (int k0 = 0; k0 < DM_; k0 += 32) {
    v16h a = load_a16(x + (size_t)arow * DM_, k0, lane);
    v16h b = load_b16(w + (size_t)col * DM_, k0, lane, true);
    acc = __builtin_amdgcn_wmma_f32_16x16x32_f16(false, a, false, b, (short)0, acc, false, false);
  }
  const int rbase = m0 + ((lane & 16) ? 8 : 0);
#pragma unroll
  for (int r = 0; r < 8; ++r) {
    const int m = rbase + r;
    const float v = acc[r];
    const int b_ = m >> 12;        // batch
    const int hw = m & 4095;       // spatial
    if (col < DI_) {
      xc_pre[((size_t)(b_ * DI_ + col) << 12) + hw] = v;       // (B,DI,H,W) for conv
    } else {
      sz[(size_t)m * DI_ + (col - DI_)] = silu(v);             // gate, (B,HW,DI)
    }
  }
}

// ---------------- K2: depthwise 3x3 SAME conv + bias + SiLU --------------------
__global__ void k_conv(const float* __restrict__ xc_pre, const float* __restrict__ cw,
                       const float* __restrict__ cb, float* __restrict__ xc_conv) {
  const int id = blockIdx.x * blockDim.x + threadIdx.x;
  if (id >= B_ * DI_ * L_) return;
  const int hw = id & 4095;
  const int c  = (id >> 12) % DI_;
  const int b  = id / (DI_ * L_);
  const int h = hw >> 6, w = hw & 63;
  const float* __restrict__ src = xc_pre + ((size_t)(b * DI_ + c) << 12);
  const float* __restrict__ wc  = cw + c * 9;
  float acc = cb[c];
#pragma unroll
  for (int dy = -1; dy <= 1; ++dy)
#pragma unroll
    for (int dx = -1; dx <= 1; ++dx) {
      const int hh = h + dy, ww = w + dx;
      if ((unsigned)hh < 64u && (unsigned)ww < 64u)
        acc += src[(hh << 6) + ww] * wc[(dy + 1) * 3 + (dx + 1)];
    }
  xc_conv[id] = silu(acc);
}

// ---------------- K3: gather 4 scan directions into u_seq (B,K,L,DI) -----------
__global__ void k_gather(const float* __restrict__ xc_conv, float* __restrict__ u_seq) {
  const size_t id = (size_t)blockIdx.x * blockDim.x + threadIdx.x;
  if (id >= (size_t)B_ * K_ * L_ * DI_) return;
  const int d = (int)(id % DI_);
  const size_t t = id / DI_;                // (b,k,l) row
  const int l = (int)(t & 4095);
  const int k = (int)((t >> 12) & 3);
  const int b = (int)(t >> 14);
  const int lm = (k & 2) ? (L_ - 1 - l) : l;
  const int s  = (k & 1) ? (((lm & 63) << 6) | (lm >> 6)) : lm;   // wh transpose
  u_seq[id] = xc_conv[((size_t)(b * DI_ + d) << 12) + s];
}

// ---------------- K4: x_proj GEMM: x_dbl = u_seq @ x_proj_w[k]^T (N padded 48) -
__global__ void k_xproj(const float* __restrict__ u_seq, const float* __restrict__ xpw,
                        float* __restrict__ x_dbl) {
  const int lane = threadIdx.x & 31;
  const int m0 = blockIdx.x * 16;            // row over (b,k,l); L%16==0 -> k uniform
  const int n0 = blockIdx.y * 16;
  const int k  = (m0 >> 12) & 3;
  const float* __restrict__ w = xpw + (size_t)k * CD_ * DI_;
  const int arow = m0 + (lane & 15);
  const int col  = n0 + (lane & 15);
  const bool valid = col < CD_;
  v8f acc = {};
#pragma unroll
  for (int k0 = 0; k0 < DI_; k0 += 32) {
    v16h a = load_a16(u_seq + (size_t)arow * DI_, k0, lane);
    v16h b = load_b16(w + (size_t)col * DI_, k0, lane, valid);
    acc = __builtin_amdgcn_wmma_f32_16x16x32_f16(false, a, false, b, (short)0, acc, false, false);
  }
  const int rbase = m0 + ((lane & 16) ? 8 : 0);
#pragma unroll
  for (int r = 0; r < 8; ++r)
    x_dbl[(size_t)(rbase + r) * CDP_ + col] = acc[r];
}

// ---------------- K5: dt projection + softplus -> delta (B,K,L,DI) -------------
__global__ void k_delta(const float* __restrict__ x_dbl, const float* __restrict__ dtw,
                        const float* __restrict__ dtb, float* __restrict__ delta) {
  const size_t id = (size_t)blockIdx.x * blockDim.x + threadIdx.x;
  if (id >= (size_t)B_ * K_ * L_ * DI_) return;
  const int d = (int)(id % DI_);
  const size_t t = id / DI_;
  const int k = (int)((t >> 12) & 3);
  const float* __restrict__ xr = x_dbl + t * CDP_;
  const float* __restrict__ wr = dtw + ((size_t)(k * DI_ + d)) * DTR_;
  float acc = dtb[k * DI_ + d];
#pragma unroll
  for (int r = 0; r < DTR_; ++r) acc += xr[r] * wr[r];
  delta[id] = (acc > 20.0f) ? acc : __logf(1.0f + __expf(acc));   // softplus
}

// ---------------- K6a: chunked scan pass 1 — per-chunk local recurrence --------
// grid = B*K*CH_ blocks; 512 threads = 16 waves; lane = (d_half, n); 3 d-passes.
// h_l = a_l*h_{l-1} + dl*u*B is linear, so each chunk is summarized by
// (h_local with h0=0, A_prod = prod a_l).
__global__ void __launch_bounds__(512, 1)
k_scan_local(const float* __restrict__ u_seq, const float* __restrict__ delta,
             const float* __restrict__ x_dbl, const float* __restrict__ A_logs,
             float* __restrict__ hc, float* __restrict__ ap) {
  const int blk = blockIdx.x;             // bk*CH_ + c
  const int bk  = blk / CH_;
  const int c   = blk % CH_;
  const int k   = bk & 3;
  const int lane = threadIdx.x & 31;
  const int wv   = threadIdx.x >> 5;      // 0..15
  const int n    = lane & 15;
  const int dh   = lane >> 4;
  int dd[3]; float Ac[3], h[3] = {0.f,0.f,0.f}, aprod[3] = {1.f,1.f,1.f};
#pragma unroll
  for (int p = 0; p < 3; ++p) {
    const int d = p * 32 + wv * 2 + dh;
    dd[p] = d;
    Ac[p] = -__expf(A_logs[((size_t)(k * DI_ + d)) * NS_ + n]);
  }
  const size_t rowbase = (size_t)bk * L_ + (size_t)c * CL_;
  for (int l = 0; l < CL_; ++l) {
    const size_t row = rowbase + l;
    if (l + 1 < CL_) {
      __builtin_prefetch(x_dbl + (row + 1) * CDP_, 0, 3);   // global_prefetch_b8
      __builtin_prefetch(delta + (row + 1) * DI_, 0, 3);
    }
    const float Bn = x_dbl[row * CDP_ + 6 + n];
    const float* __restrict__ ur = u_seq + row * DI_;
    const float* __restrict__ dr = delta + row * DI_;
#pragma unroll
    for (int p = 0; p < 3; ++p) {
      const float u  = ur[dd[p]];
      const float dl = dr[dd[p]];
      const float a  = __expf(dl * Ac[p]);
      h[p] = h[p] * a + (dl * u) * Bn;
      aprod[p] *= a;
    }
  }
#pragma unroll
  for (int p = 0; p < 3; ++p) {
    const size_t idx = (((size_t)bk * CH_ + c) * DI_ + dd[p]) * NS_ + n;
    hc[idx] = h[p];
    ap[idx] = aprod[p];
  }
}

// ---------------- K6b: chunked scan pass 2 — exclusive scan over chunk carries -
// one thread per (bk,d,n): 32-step serial scan; hc[] rewritten in place with the
// state ENTERING each chunk.
__global__ void k_scan_carry(float* __restrict__ hc, const float* __restrict__ ap) {
  const int id = blockIdx.x * blockDim.x + threadIdx.x;   // B*K*DI*NS = 24576
  if (id >= B_ * K_ * DI_ * NS_) return;
  const int bk = id / (DI_ * NS_);
  const int dn = id % (DI_ * NS_);
  float H = 0.f;
#pragma unroll 4
  for (int c = 0; c < CH_; ++c) {
    const size_t idx = ((size_t)bk * CH_ + c) * (DI_ * NS_) + dn;
    const float hl = hc[idx];
    const float a  = ap[idx];
    hc[idx] = H;                // incoming state for chunk c
    H = a * H + hl;
  }
}

// ---------------- K6c: chunked scan pass 3 — seeded re-run, emit y -------------
__global__ void __launch_bounds__(512, 1)
k_scan_final(const float* __restrict__ u_seq, const float* __restrict__ delta,
             const float* __restrict__ x_dbl, const float* __restrict__ A_logs,
             const float* __restrict__ Dsk, const float* __restrict__ hc,
             float* __restrict__ yseq) {
  const int blk = blockIdx.x;             // bk*CH_ + c
  const int bk  = blk / CH_;
  const int c   = blk % CH_;
  const int k   = bk & 3;
  const int lane = threadIdx.x & 31;
  const int wv   = threadIdx.x >> 5;
  const int n    = lane & 15;
  const int dh   = lane >> 4;
  int dd[3]; float Ac[3], Dc[3], h[3];
#pragma unroll
  for (int p = 0; p < 3; ++p) {
    const int d = p * 32 + wv * 2 + dh;
    dd[p] = d;
    Ac[p] = -__expf(A_logs[((size_t)(k * DI_ + d)) * NS_ + n]);
    Dc[p] = Dsk[k * DI_ + d];
    h[p]  = hc[(((size_t)bk * CH_ + c) * DI_ + d) * NS_ + n];   // seeded state
  }
  const size_t rowbase = (size_t)bk * L_ + (size_t)c * CL_;
  for (int l = 0; l < CL_; ++l) {
    const size_t row = rowbase + l;
    if (l + 1 < CL_) {
      __builtin_prefetch(x_dbl + (row + 1) * CDP_, 0, 3);
      __builtin_prefetch(delta + (row + 1) * DI_, 0, 3);
    }
    const float* __restrict__ xd = x_dbl + row * CDP_;
    const float Bn = xd[6 + n];
    const float Cn = xd[22 + n];
    const float* __restrict__ ur = u_seq + row * DI_;
    const float* __restrict__ dr = delta + row * DI_;
#pragma unroll
    for (int p = 0; p < 3; ++p) {
      const float u  = ur[dd[p]];
      const float dl = dr[dd[p]];
      const float a  = __expf(dl * Ac[p]);
      h[p] = h[p] * a + (dl * u) * Bn;
      float yc = h[p] * Cn;
      yc += __shfl_xor(yc, 1, 32);
      yc += __shfl_xor(yc, 2, 32);
      yc += __shfl_xor(yc, 4, 32);
      yc += __shfl_xor(yc, 8, 32);
      if (n == 0) yseq[row * DI_ + dd[p]] = yc + Dc[p] * u;      // fold D*u skip
    }
  }
}

// ---------------- K7: merge 4 directions + LayerNorm + SiLU gate ---------------
__global__ void k_combine(const float* __restrict__ yseq, const float* __restrict__ sz,
                          const float* __restrict__ nw, const float* __restrict__ nb,
                          float* __restrict__ g) {
  const int lane = threadIdx.x & 31;
  const int wv   = threadIdx.x >> 5;                 // 0..3
  const int row  = blockIdx.x * 4 + wv;              // b*4096 + p
  const int b = row >> 12;
  const int p = row & 4095;
  const int p1 = ((p & 63) << 6) | (p >> 6);         // wh2hw index
  const size_t base = (size_t)b * K_ * L_;
  const size_t r0 = (base + 0 * L_ + p)             * DI_;
  const size_t r1 = (base + 1 * L_ + p1)            * DI_;
  const size_t r2 = (base + 2 * L_ + (L_ - 1 - p))  * DI_;
  const size_t r3 = (base + 3 * L_ + (L_ - 1 - p1)) * DI_;
  float v[3], s1 = 0.f, s2 = 0.f;
#pragma unroll
  for (int j = 0; j < 3; ++j) {
    const int d = lane + j * 32;
    const float y = yseq[r0 + d] + yseq[r1 + d] + yseq[r2 + d] + yseq[r3 + d];
    v[j] = y; s1 += y; s2 += y * y;
  }
#pragma unroll
  for (int m = 1; m < 32; m <<= 1) {
    s1 += __shfl_xor(s1, m, 32);
    s2 += __shfl_xor(s2, m, 32);
  }
  const float mu  = s1 * (1.0f / DI_);
  const float var = s2 * (1.0f / DI_) - mu * mu;
  const float inv = rsqrtf(var + 1e-5f);
#pragma unroll
  for (int j = 0; j < 3; ++j) {
    const int d = lane + j * 32;
    const float yn = (v[j] - mu) * inv * nw[d] + nb[d];
    g[(size_t)row * DI_ + d] = yn * sz[(size_t)row * DI_ + d];
  }
}

// ---------------- K8: out_proj GEMM: out = g @ Wout^T --------------------------
__global__ void k_out_proj(const float* __restrict__ g, const float* __restrict__ w,
                           float* __restrict__ out) {
  const int lane = threadIdx.x & 31;
  const int m0 = blockIdx.x * 16;
  const int n0 = blockIdx.y * 16;
  const int arow = m0 + (lane & 15);
  const int col  = n0 + (lane & 15);
  v8f acc = {};
#pragma unroll
  for (int k0 = 0; k0 < DI_; k0 += 32) {
    v16h a = load_a16(g + (size_t)arow * DI_, k0, lane);
    v16h b = load_b16(w + (size_t)col * DI_, k0, lane, true);
    acc = __builtin_amdgcn_wmma_f32_16x16x32_f16(false, a, false, b, (short)0, acc, false, false);
  }
  const int rbase = m0 + ((lane & 16) ? 8 : 0);
#pragma unroll
  for (int r = 0; r < 8; ++r)
    out[(size_t)(rbase + r) * DM_ + col] = acc[r];
}

// ---------------- launch -------------------------------------------------------
extern "C" void kernel_launch(void* const* d_in, const int* in_sizes, int n_in,
                              void* d_out, int out_size, void* d_ws, size_t ws_size,
                              hipStream_t stream) {
  (void)in_sizes; (void)n_in; (void)out_size; (void)ws_size;
  const float* x    = (const float*)d_in[0];
  const float* ipw  = (const float*)d_in[1];
  const float* cw   = (const float*)d_in[2];
  const float* cb   = (const float*)d_in[3];
  const float* xpw  = (const float*)d_in[4];
  const float* dtw  = (const float*)d_in[5];
  const float* dtb  = (const float*)d_in[6];
  const float* alog = (const float*)d_in[7];
  const float* Dsk  = (const float*)d_in[8];
  const float* nw   = (const float*)d_in[9];
  const float* nb   = (const float*)d_in[10];
  const float* opw  = (const float*)d_in[11];
  float* out = (float*)d_out;
  float* ws  = (float*)d_ws;

  size_t o = 0;
  float* xc_pre  = ws + o; o += (size_t)B_ * DI_ * L_;            // 1.57M
  float* sz      = ws + o; o += (size_t)B_ * L_ * DI_;            // 1.57M
  float* xc_conv = ws + o; o += (size_t)B_ * DI_ * L_;            // 1.57M
  float* u_seq   = ws + o; o += (size_t)B_ * K_ * L_ * DI_;       // 6.29M
  float* x_dbl   = ws + o; o += (size_t)B_ * K_ * L_ * CDP_;      // 3.15M
  float* delta   = ws + o; o += (size_t)B_ * K_ * L_ * DI_;       // 6.29M
  float* yseq    = ws + o; o += (size_t)B_ * K_ * L_ * DI_;       // 6.29M
  float* hc      = ws + o; o += (size_t)B_ * K_ * CH_ * DI_ * NS_; // 0.79M
  float* ap      = ws + o; o += (size_t)B_ * K_ * CH_ * DI_ * NS_; // 0.79M
  float* g       = xc_pre;                                        // reuse (free by K7)

  k_in_proj   <<<dim3((B_ * L_) / 16, (2 * DI_) / 16), 32, 0, stream>>>(x, ipw, xc_pre, sz);
  k_conv      <<<(B_ * DI_ * L_ + 255) / 256, 256, 0, stream>>>(xc_pre, cw, cb, xc_conv);
  k_gather    <<<(B_ * K_ * L_ * DI_ + 255) / 256, 256, 0, stream>>>(xc_conv, u_seq);
  k_xproj     <<<dim3((B_ * K_ * L_) / 16, CDP_ / 16), 32, 0, stream>>>(u_seq, xpw, x_dbl);
  k_delta     <<<(B_ * K_ * L_ * DI_ + 255) / 256, 256, 0, stream>>>(x_dbl, dtw, dtb, delta);
  k_scan_local<<<B_ * K_ * CH_, 512, 0, stream>>>(u_seq, delta, x_dbl, alog, hc, ap);
  k_scan_carry<<<(B_ * K_ * DI_ * NS_ + 255) / 256, 256, 0, stream>>>(hc, ap);
  k_scan_final<<<B_ * K_ * CH_, 512, 0, stream>>>(u_seq, delta, x_dbl, alog, Dsk, hc, yseq);
  k_combine   <<<(B_ * L_) / 4, 128, 0, stream>>>(yseq, sz, nw, nb, g);
  k_out_proj  <<<dim3((B_ * L_) / 16, DM_ / 16), 32, 0, stream>>>(g, opw, out);
}
